// AutoEncoder_64854006170336
// MI455X (gfx1250) — matrix operand order repaired and checked by reading, not compiled, not run on hardware
//
#include <hip/hip_runtime.h>
#include <hip/hip_bf16.h>

// ---------------- model constants ----------------
#define BB      1024
#define JJ      24
#define AA      25          // tokens = joints + cls
#define DM      512
#define NH      8
#define DEPTH   64
#define NL      6
#define DIN     64

typedef __bf16 bf16;
typedef __bf16 v8bf  __attribute__((ext_vector_type(8)));
typedef __bf16 v16bf __attribute__((ext_vector_type(16)));
typedef float  v8f   __attribute__((ext_vector_type(8)));
typedef int    v4i   __attribute__((ext_vector_type(4)));

union FragU { v16bf v; v8bf h[2]; };

// LDS tile row stride (bf16 elements): 32 payload + 8 pad -> 80B rows,
// start banks 20*i mod 64 are distinct multiples of 4 => conflict-free b128.
#define LDS_STRIDE 40

// ---- CDNA5 async global->LDS path (guarded; falls back to ds_store) ----
#if defined(__has_builtin)
#if __has_builtin(__builtin_amdgcn_global_load_async_to_lds_b128) && \
    __has_builtin(__builtin_amdgcn_s_wait_asynccnt)
#define USE_ASYNC_LDS 1
#endif
#endif
#ifndef USE_ASYNC_LDS
#define USE_ASYNC_LDS 0
#endif

__device__ __forceinline__ void copy16(const bf16* g, bf16* l)
{
#if USE_ASYNC_LDS
    // builtin signature (from hipcc diagnostic): (v4i AS1*, v4i AS3*, imm, imm)
    __builtin_amdgcn_global_load_async_to_lds_b128(
        (__attribute__((address_space(1))) v4i*)(unsigned long long)g,
        (__attribute__((address_space(3))) v4i*)(unsigned int)(unsigned long long)l,
        0, 0);
#else
    *(v8bf*)l = *(const v8bf*)g;
#endif
}

__device__ __forceinline__ void wait_async_lds()
{
#if USE_ASYNC_LDS
    __builtin_amdgcn_s_wait_asynccnt(0);
#endif
}

// ---------------------------------------------------------------------------
// Transpose 512x512 fp32 matrix -> bf16 W^T (Wt[n*512+k] = W[k*512+n]).
// One matrix per blockIdx.z, 32x32 tile per block.
// ---------------------------------------------------------------------------
__global__ void __launch_bounds__(256)
transpose_to_bf16_kernel(const float* __restrict__ W, bf16* __restrict__ Wt)
{
    __shared__ float tile[32][33];
    const long  mz  = (long)blockIdx.z * DM * DM;
    const float* src = W  + mz;
    bf16*        dst = Wt + mz;
    const int r0 = blockIdx.y * 32;
    const int c0 = blockIdx.x * 32;
    const int tr = threadIdx.x >> 5;   // 0..7
    const int tc = threadIdx.x & 31;   // 0..31
#pragma unroll
    for (int i = 0; i < 32; i += 8)
        tile[tr + i][tc] = src[(long)(r0 + tr + i) * DM + (c0 + tc)];
    __syncthreads();
#pragma unroll
    for (int i = 0; i < 32; i += 8)
        dst[(long)(c0 + tr + i) * DM + (r0 + tc)] = (bf16)tile[tc][tr + i];
}

// ---------------------------------------------------------------------------
// Input embedding: h[b,0,:]=cls ; h[b,1+j,:]= x[b,j,:] @ We[j] + be[j]
// Writes h in f32 (residual path) and bf16 (WMMA operand path).
// ---------------------------------------------------------------------------
__global__ void __launch_bounds__(256)
embed_kernel(const float* __restrict__ x,  const float* __restrict__ We,
             const float* __restrict__ be, const float* __restrict__ cls,
             float* __restrict__ hf, bf16* __restrict__ hb)
{
    const long row = blockIdx.x;              // 0 .. B*A-1
    const int  b   = (int)(row / AA);
    const int  a   = (int)(row % AA);
    const int  t   = threadIdx.x;
    float* out  = hf + row * DM;
    bf16*  outb = hb + row * DM;

    if (a == 0) {
        for (int c = t; c < DM; c += 256) {
            float v = cls[c];
            out[c] = v; outb[c] = (bf16)v;
        }
        return;
    }
    const int j = a - 1;
    __shared__ float xs[DIN];
    if (t < DIN) xs[t] = x[((long)b * JJ + j) * DIN + t];
    __syncthreads();
    const float* w  = We + (long)j * DIN * DM;
    const float* bj = be + (long)j * DM;
    for (int c = t; c < DM; c += 256) {
        float s = bj[c];
#pragma unroll
        for (int kk = 0; kk < DIN; ++kk)
            s += xs[kk] * w[(long)kk * DM + c];
        out[c] = s; outb[c] = (bf16)s;
    }
}

// ---------------------------------------------------------------------------
// bf16 WMMA GEMM: C(f32) = A(bf16)[M x 512] x Bt(bf16, pre-transposed) + bias
// Block = 256 threads (8 waves), tile 128(M) x 256(N), K staged by 32 through
// double-buffered LDS filled with async global->LDS (ASYNCcnt) when available.
// Wave tile 64x64: 4 A-frags x 4 B-frags -> 16 v_wmma per K step.
// blockIdx.z batches independent slices (25 per-joint query projections).
// ---------------------------------------------------------------------------
__global__ void __launch_bounds__(256)
gemm_bf16_wmma_kernel(const bf16*  __restrict__ A,  long lda,  long a_zstride,
                      const bf16*  __restrict__ Bt, long b_zstride,
                      const float* __restrict__ bias, long bias_zstride,
                      float* __restrict__ C, long ldc, long c_zstride)
{
    const int tid  = threadIdx.x;
    const int lane = tid & 31;
    const int wid  = tid >> 5;

    A    += (long)blockIdx.z * a_zstride;
    Bt   += (long)blockIdx.z * b_zstride;
    bias += (long)blockIdx.z * bias_zstride;
    C    += (long)blockIdx.z * c_zstride;

    const long tileM = (long)blockIdx.y * 128;
    const int  tileN = blockIdx.x * 256;

    __shared__ bf16 ldsA[2][128 * LDS_STRIDE];   // 2 x 10240 B
    __shared__ bf16 ldsB[2][256 * LDS_STRIDE];   // 2 x 20480 B

    const int waveM = (wid & 1) * 64;   // 0,64
    const int waveN = (wid >> 1) * 64;  // 0,64,128,192

    v8f acc[4][4] = {};

    const int lr = tid >> 2;            // 0..63
    const int lk = (tid & 3) * 8;       // 0,8,16,24 (8 bf16 = 16B)

    // tile loader: A 128x32, B^T 256x32
    auto load_tiles = [&](int k0, bf16* la, bf16* lb) {
        const bf16* ga = A + (tileM + lr) * lda + (k0 + lk);
        copy16(ga,            &la[lr * LDS_STRIDE + lk]);
        copy16(ga + 64 * lda, &la[(lr + 64) * LDS_STRIDE + lk]);
        const bf16* gb = Bt + (long)(tileN + lr) * DM + (k0 + lk);
#pragma unroll
        for (int r = 0; r < 256; r += 64)
            copy16(gb + (long)r * DM, &lb[(lr + r) * LDS_STRIDE + lk]);
    };

    load_tiles(0, ldsA[0], ldsB[0]);          // prologue
    int cur = 0;

    for (int k0 = 0; k0 < DM; k0 += 32) {
        wait_async_lds();
        __syncthreads();
        if (k0 + 32 < DM)
            load_tiles(k0 + 32, ldsA[cur ^ 1], ldsB[cur ^ 1]);

        const bf16* la = ldsA[cur];
        const bf16* lb = ldsB[cur];

        // ---- fragment assembly (ISA 7.12.2 layouts) ----
        const int r16  = lane & 15;
        const int half = lane >> 4;
        v16bf afrag[4], bfrag[4];
#pragma unroll
        for (int i = 0; i < 4; ++i) {
            const bf16* p = &la[(waveM + i * 16 + r16) * LDS_STRIDE + half * 8];
            FragU u;
            u.h[0] = *(const v8bf*)p;         // K = half*8 .. +7
            u.h[1] = *(const v8bf*)(p + 16);  // K = 16 + half*8 .. +7
            afrag[i] = u.v;
        }
#pragma unroll
        for (int j = 0; j < 4; ++j) {
            const bf16* p = &lb[(waveN + j * 16 + r16) * LDS_STRIDE + half * 8];
            FragU u;
            u.h[0] = *(const v8bf*)p;         // K = half*8 .. +7
            u.h[1] = *(const v8bf*)(p + 16);  // K = 16 + half*8 .. +7
            bfrag[j] = u.v;
        }
        // ---- 16x v_wmma_f32_16x16x32_bf16 ----
#pragma unroll
        for (int i = 0; i < 4; ++i)
#pragma unroll
            for (int j = 0; j < 4; ++j)
                acc[i][j] = __builtin_amdgcn_wmma_f32_16x16x32_bf16(
                    false, afrag[i], false, bfrag[j],
                    (short)0, acc[i][j], false, false);

        cur ^= 1;
    }

    // ---- epilogue: bias add + store ----
    const int cn = lane & 15;
    const int mh = (lane >> 4) * 8;
#pragma unroll
    for (int j = 0; j < 4; ++j) {
        const int col = tileN + waveN + j * 16 + cn;
        const float bv = bias[col];
#pragma unroll
        for (int i = 0; i < 4; ++i) {
#pragma unroll
            for (int e = 0; e < 8; ++e) {
                const long rowg = tileM + waveM + i * 16 + mh + e;
                C[rowg * ldc + col] = acc[i][j][e] + bv;
            }
        }
    }
}

// ---------------------------------------------------------------------------
// Attention: per (b, head) wave32. A=25 tokens, depth=64, fp32 softmax.
// Writes o in bf16 (operand of the Wo WMMA GEMM).
// ---------------------------------------------------------------------------
__global__ void __launch_bounds__(32)
attn_kernel(const float* __restrict__ q, const float* __restrict__ k,
            const float* __restrict__ v, bf16* __restrict__ o)
{
    const int bh   = blockIdx.x;      // 0 .. B*H-1
    const int b    = bh >> 3;
    const int h    = bh & 7;
    const int lane = threadIdx.x;
    __shared__ float ks[AA][DEPTH];
    __shared__ float vs[AA][DEPTH];
    const long base = (long)b * AA * DM + h * DEPTH;
    for (int idx = lane; idx < AA * DEPTH; idx += 32) {
        const int a = idx >> 6, d = idx & 63;
        ks[a][d] = k[base + (long)a * DM + d];
        vs[a][d] = v[base + (long)a * DM + d];
    }
    __syncthreads();
    if (lane >= AA) return;

    float qr[DEPTH];
    const float* qp = q + base + (long)lane * DM;
#pragma unroll
    for (int d = 0; d < DEPTH; ++d) qr[d] = qp[d];

    float sc[AA];
    float mx = -1e30f;
#pragma unroll
    for (int a = 0; a < AA; ++a) {
        float s = 0.f;
#pragma unroll
        for (int d = 0; d < DEPTH; ++d) s += qr[d] * ks[a][d];
        s *= 0.125f;                 // 1/sqrt(depth=64)
        sc[a] = s;
        mx = fmaxf(mx, s);
    }
    float sum = 0.f;
#pragma unroll
    for (int a = 0; a < AA; ++a) { sc[a] = __expf(sc[a] - mx); sum += sc[a]; }
    const float inv = 1.f / sum;

    bf16* op = o + base + (long)lane * DM;
#pragma unroll
    for (int d = 0; d < DEPTH; ++d) {
        float acc = 0.f;
#pragma unroll
        for (int a = 0; a < AA; ++a) acc += sc[a] * vs[a][d];
        op[d] = (bf16)(acc * inv);
    }
}

// ---------------------------------------------------------------------------
// Residual + LayerNorm over D=512; one (b,a) row per block of 256.
// Rewrites h in f32 + bf16; optionally copies to final output.
// ---------------------------------------------------------------------------
__global__ void __launch_bounds__(256)
resid_ln_kernel(const float* __restrict__ hin, const float* __restrict__ proj,
                const float* __restrict__ g, const float* __restrict__ bta,
                float* __restrict__ hf, bf16* __restrict__ hb,
                float* __restrict__ final_out)
{
    const long row = blockIdx.x;
    const int  t   = threadIdx.x;
    __shared__ float red[256];

    const float a0 = hin[row * DM + t]       + proj[row * DM + t];
    const float a1 = hin[row * DM + t + 256] + proj[row * DM + t + 256];
    red[t] = a0 + a1;
    __syncthreads();
    for (int s = 128; s > 0; s >>= 1) {
        if (t < s) red[t] += red[t + s];
        __syncthreads();
    }
    const float mean = red[0] * (1.f / DM);
    __syncthreads();
    const float d0 = a0 - mean, d1 = a1 - mean;
    red[t] = d0 * d0 + d1 * d1;
    __syncthreads();
    for (int s = 128; s > 0; s >>= 1) {
        if (t < s) red[t] += red[t + s];
        __syncthreads();
    }
    const float rstd = rsqrtf(red[0] * (1.f / DM) + 1e-5f);

    const float o0 = d0 * rstd * g[t]       + bta[t];
    const float o1 = d1 * rstd * g[t + 256] + bta[t + 256];
    hf[row * DM + t]       = o0;
    hf[row * DM + t + 256] = o1;
    hb[row * DM + t]       = (bf16)o0;
    hb[row * DM + t + 256] = (bf16)o1;
    if (final_out) {
        final_out[row * DM + t]       = o0;
        final_out[row * DM + t + 256] = o1;
    }
}

// ---------------------------------------------------------------------------
// host launcher
// ---------------------------------------------------------------------------
extern "C" void kernel_launch(void* const* d_in, const int* in_sizes, int n_in,
                              void* d_out, int out_size, void* d_ws, size_t ws_size,
                              hipStream_t stream)
{
    (void)in_sizes; (void)n_in; (void)out_size; (void)ws_size;

    const float* x    = (const float*)d_in[0];
    const float* We   = (const float*)d_in[1];
    const float* be   = (const float*)d_in[2];
    const float* cls  = (const float*)d_in[3];
    const float* Wk   = (const float*)d_in[4];
    const float* bk   = (const float*)d_in[5];
    const float* Wv   = (const float*)d_in[6];
    const float* bv   = (const float*)d_in[7];
    const float* Wq   = (const float*)d_in[8];
    const float* bq   = (const float*)d_in[9];
    const float* Wo   = (const float*)d_in[10];
    const float* bo   = (const float*)d_in[11];
    const float* lng  = (const float*)d_in[12];
    const float* lnb  = (const float*)d_in[13];
    float*       out  = (float*)d_out;

    // ---- workspace carve-up (deterministic) ----
    char*  ws  = (char*)d_ws;
    size_t off = 0;
    auto carve = [&](size_t bytes) -> void* {
        void* p = ws + off;
        off += (bytes + 255) & ~(size_t)255;
        return p;
    };
    const size_t MAT  = (size_t)DM * DM;                  // 512*512
    bf16*  WkT  = (bf16*) carve(NL * MAT * sizeof(bf16));
    bf16*  WvT  = (bf16*) carve(NL * MAT * sizeof(bf16));
    bf16*  WoT  = (bf16*) carve(NL * MAT * sizeof(bf16));
    bf16*  WqT  = (bf16*) carve((size_t)NL * AA * MAT * sizeof(bf16));
    const size_t ROWS = (size_t)BB * AA;                  // 25600
    float* hf   = (float*)carve(ROWS * DM * sizeof(float));
    bf16*  hb   = (bf16*) carve(ROWS * DM * sizeof(bf16));
    float* qf   = (float*)carve(ROWS * DM * sizeof(float));
    float* kf   = (float*)carve(ROWS * DM * sizeof(float));
    float* vf   = (float*)carve(ROWS * DM * sizeof(float));
    bf16*  ob   = (bf16*) carve(ROWS * DM * sizeof(bf16));
    float* pj   = (float*)carve(ROWS * DM * sizeof(float));

    // ---- weight prep: fp32 -> bf16 transpose ----
    transpose_to_bf16_kernel<<<dim3(16, 16, NL),      256, 0, stream>>>(Wk, WkT);
    transpose_to_bf16_kernel<<<dim3(16, 16, NL),      256, 0, stream>>>(Wv, WvT);
    transpose_to_bf16_kernel<<<dim3(16, 16, NL),      256, 0, stream>>>(Wo, WoT);
    transpose_to_bf16_kernel<<<dim3(16, 16, NL * AA), 256, 0, stream>>>(Wq, WqT);

    // ---- embedding ----
    embed_kernel<<<dim3((unsigned)ROWS), 256, 0, stream>>>(x, We, be, cls, hf, hb);

    const dim3 gemmBig(DM / 256, (unsigned)(ROWS / 128), 1);   // 2 x 200
    const dim3 gemmQ  (DM / 256, BB / 128, AA);                // 2 x 8 x 25

    for (int l = 0; l < NL; ++l) {
        // K = h @ Wk[l] + bk[l]
        gemm_bf16_wmma_kernel<<<gemmBig, 256, 0, stream>>>(
            hb, DM, 0, WkT + (size_t)l * MAT, 0,
            bk + (size_t)l * DM, 0, kf, DM, 0);
        // V = h @ Wv[l] + bv[l]
        gemm_bf16_wmma_kernel<<<gemmBig, 256, 0, stream>>>(
            hb, DM, 0, WvT + (size_t)l * MAT, 0,
            bv + (size_t)l * DM, 0, vf, DM, 0);
        // Q: per-joint projection, blockIdx.z = token a
        gemm_bf16_wmma_kernel<<<gemmQ, 256, 0, stream>>>(
            hb, (long)AA * DM, DM,
            WqT + (size_t)l * AA * MAT, (long)MAT,
            bq + (size_t)l * AA * DM, DM,
            qf, (long)AA * DM, DM);
        // attention
        attn_kernel<<<dim3(BB * NH), 32, 0, stream>>>(qf, kf, vf, ob);
        // O = attn @ Wo[l] + bo[l]
        gemm_bf16_wmma_kernel<<<gemmBig, 256, 0, stream>>>(
            ob, DM, 0, WoT + (size_t)l * MAT, 0,
            bo + (size_t)l * DM, 0, pj, DM, 0);
        // h = LN(h + O)
        resid_ln_kernel<<<dim3((unsigned)ROWS), 256, 0, stream>>>(
            hf, pj, lng + (size_t)l * DM, lnb + (size_t)l * DM,
            hf, hb, (l == NL - 1) ? out : nullptr);
    }
}
